// SoftEmbedding_77378130804767
// MI455X (gfx1250) — compile-verified, matrix operand order checked by reading, not run on hardware
//
#include <hip/hip_runtime.h>

typedef __attribute__((ext_vector_type(16))) __bf16 v16bf;
typedef __attribute__((ext_vector_type(8)))  float  v8f;

#define NTOK    (4096 * 200)          // B*S tokens
#define NEMB    512
#define EDIM    64
#define KSTEP   32
#define NSTEPS  (NEMB / KSTEP)        // 16
#define WAVES   8
#define TPB     (WAVES * 32)          // 256 threads
#define TOKS_PER_TILE 16
#define NTILES  (NTOK / TOKS_PER_TILE) // 51200
#define LOG2E   1.4426950408889634f

// K offset pattern for 16-bit A/B fragments (wave32):
// lanes 0-15 own K%32 in {0..7,16..23}, lanes 16-31 own {8..15,24..31},
// packed two-per-dword across 8 VGPRs.
__device__ __forceinline__ int koff(int half, int v) {
  return (v < 4) ? (half * 8 + 2 * v) : (16 + half * 8 + 2 * (v - 4));
}

__device__ __forceinline__ unsigned f2bf(float f) {
  union { float f; unsigned u; } c; c.f = f;
  unsigned r = c.u + 0x7FFFu + ((c.u >> 16) & 1u);   // round to nearest even
  return r >> 16;
}

__global__ __launch_bounds__(TPB) void soft_emb_wmma_kernel(
    const float* __restrict__ xin,   // [NTOK]
    const float* __restrict__ pw,    // [512]
    const float* __restrict__ pb,    // [512]
    const float* __restrict__ emb,   // [512][64]
    float* __restrict__ out) {       // [NTOK][64]
  // B fragments, bf16, pre-swizzled to WMMA lane layout: [step][ntile][lane][8 dwords]
  __shared__ __align__(32) unsigned ldsB[NSTEPS * 4 * 32 * 8];   // 64 KB
  // (w*log2e, b*log2e) pairs in A-fragment K order: [step][half][8 x float4]
  __shared__ __align__(16) float    ldsWB[NSTEPS * 2 * 8 * 4];   // 4 KB

  const int tid = threadIdx.x;

  // ---- one-time staging: emb f32 -> bf16 fragments in LDS ----
  for (int idx = tid; idx < NSTEPS * 4 * 32 * 8; idx += TPB) {
    int f = idx >> 8;              // fragment id = s*4 + nt (256 dwords each)
    int r = idx & 255;
    int l = r >> 3;                // lane 0..31
    int v = r & 7;                 // dword 0..7
    int s = f >> 2, nt = f & 3;
    int half = l >> 4;
    int n = nt * 16 + (l & 15);
    int kb = s * KSTEP + koff(half, v);
    unsigned lo = f2bf(emb[kb * EDIM + n]);
    unsigned hi = f2bf(emb[(kb + 1) * EDIM + n]);
    ldsB[idx] = lo | (hi << 16);
  }
  // ---- one-time staging: log2e-scaled proj weights/bias ----
  for (int t = tid; t < NSTEPS * 2 * 8; t += TPB) {
    int s = t >> 4, half = (t >> 3) & 1, v = t & 7;
    int kb = s * KSTEP + koff(half, v);
    ldsWB[t * 4 + 0] = pw[kb] * LOG2E;
    ldsWB[t * 4 + 1] = pb[kb] * LOG2E;
    ldsWB[t * 4 + 2] = pw[kb + 1] * LOG2E;
    ldsWB[t * 4 + 3] = pb[kb + 1] * LOG2E;
  }
  __syncthreads();

  const int wave = tid >> 5;
  const int lane = tid & 31;
  const int half = lane >> 4;
  const int col  = lane & 15;

  v16bf ones;
  #pragma unroll
  for (int i = 0; i < 16; ++i) ones[i] = (__bf16)1.0f;

  const float4* wb4 = (const float4*)ldsWB;

  for (int tile = blockIdx.x * WAVES + wave; tile < NTILES;
       tile += gridDim.x * WAVES) {
    const int tok0 = tile * TOKS_PER_TILE;
    const float x = xin[tok0 + col];   // lane's token row (dup across halves)

    v8f acc0 = {}, acc1 = {}, acc2 = {}, acc3 = {}, accS = {};

    // Unroll-2 only: enough to pipeline ds_loads against WMMAs without
    // blowing past 256 VGPRs (full unroll caused scratch spills).
    #pragma unroll 2
    for (int s = 0; s < NSTEPS; ++s) {
      // Build A fragment in registers: e = 2^(x*w' + b'), bf16-packed.
      const float4* wb = &wb4[s * 16 + half * 8];
      v16bf a;
      #pragma unroll
      for (int v = 0; v < 8; ++v) {
        float4 q = wb[v];  // w'[k], b'[k], w'[k+1], b'[k+1]
        float e0 = __builtin_amdgcn_exp2f(fmaf(x, q.x, q.y));
        float e1 = __builtin_amdgcn_exp2f(fmaf(x, q.z, q.w));
        a[2 * v]     = (__bf16)e0;
        a[2 * v + 1] = (__bf16)e1;
      }
      // B fragments: 2x ds_load_b128 each, fragment stride = 256 dwords = 32 v16bf
      const v16bf* bbase = (const v16bf*)&ldsB[(s * 4) * 256 + lane * 8];
      v16bf b0 = bbase[0];
      v16bf b1 = bbase[32];
      v16bf b2 = bbase[64];
      v16bf b3 = bbase[96];

      acc0 = __builtin_amdgcn_wmma_f32_16x16x32_bf16(false, a, false, b0,
                                                     (short)0, acc0, false, false);
      acc1 = __builtin_amdgcn_wmma_f32_16x16x32_bf16(false, a, false, b1,
                                                     (short)0, acc1, false, false);
      acc2 = __builtin_amdgcn_wmma_f32_16x16x32_bf16(false, a, false, b2,
                                                     (short)0, acc2, false, false);
      acc3 = __builtin_amdgcn_wmma_f32_16x16x32_bf16(false, a, false, b3,
                                                     (short)0, acc3, false, false);
      // softmax denominator via WMMA against all-ones B
      accS = __builtin_amdgcn_wmma_f32_16x16x32_bf16(false, a, false, ones,
                                                     (short)0, accS, false, false);
    }

    // C/D layout: VGPR j -> row j (lanes 0-15) / row j+8 (lanes 16-31), N = col
    float* orow = out + (size_t)(tok0 + half * 8) * EDIM + col;
    #pragma unroll
    for (int j = 0; j < 8; ++j) {
      float inv = __builtin_amdgcn_rcpf(accS[j]);
      orow[(size_t)j * EDIM +  0] = acc0[j] * inv;
      orow[(size_t)j * EDIM + 16] = acc1[j] * inv;
      orow[(size_t)j * EDIM + 32] = acc2[j] * inv;
      orow[(size_t)j * EDIM + 48] = acc3[j] * inv;
    }
  }
}

extern "C" void kernel_launch(void* const* d_in, const int* in_sizes, int n_in,
                              void* d_out, int out_size, void* d_ws, size_t ws_size,
                              hipStream_t stream) {
  const float* xin = (const float*)d_in[0];  // input_numeric [B,S,1]
  const float* pw  = (const float*)d_in[1];  // proj_w [512,1]
  const float* pb  = (const float*)d_in[2];  // proj_b [512]
  const float* emb = (const float*)d_in[3];  // emb_table [512,64]
  float* out = (float*)d_out;                // [B,S,64] f32

  dim3 grid(2048), block(TPB);
  hipLaunchKernelGGL(soft_emb_wmma_kernel, grid, block, 0, stream,
                     xin, pw, pb, emb, out);
}
